// GraphTransformerClassifier_83588653515022
// MI455X (gfx1250) — matrix operand order repaired and checked by reading, not compiled, not run on hardware
//
#include <hip/hip_runtime.h>
#include <hip/hip_bf16.h>
#include <math.h>

// Problem constants (match reference)
#define N_NODES   20000
#define N_EDGES   200000
#define H_HEADS   4
#define D_DIM     64
#define F_INPUT   128
#define EDGE_DIM  16
#define G_GRAPHS  256
#define N_CLASSES 10
#define HID       64
#define HD        256            // H*D
#define SCALE_ATT 0.125f         // 1/sqrt(64)
#define ENC_NEG_INF 0x007FFFFFu  // ordered-uint encoding of -inf

typedef __attribute__((ext_vector_type(16))) __bf16 v16bf;
typedef __attribute__((ext_vector_type(8)))  float  v8f;

__device__ __forceinline__ __bf16 f2bf(float f) {
  // round-to-nearest-even f32 -> bf16
  unsigned u = __float_as_uint(f);
  unsigned r = (u + 0x7FFFu + ((u >> 16) & 1u)) >> 16;
  unsigned short s = (unsigned short)r;
  __bf16 o; __builtin_memcpy(&o, &s, 2); return o;
}
__device__ __forceinline__ unsigned fenc(float f) {  // order-preserving float->uint
  unsigned u = __float_as_uint(f);
  return (u & 0x80000000u) ? ~u : (u | 0x80000000u);
}
__device__ __forceinline__ float fdec(unsigned u) {
  return __uint_as_float((u & 0x80000000u) ? (u & 0x7FFFFFFFu) : ~u);
}

// ---------------------------------------------------------------------------
// Generic buffer fill
// ---------------------------------------------------------------------------
__global__ void k_fill_u32(unsigned* __restrict__ p, unsigned v, long long n) {
  long long t = (long long)blockIdx.x * blockDim.x + threadIdx.x;
  if (t < n) p[t] = v;
}

// ---------------------------------------------------------------------------
// Pack f32 row-major A[M,K] into WMMA A-fragment order, bf16.
// Layout: ((tm*KB + kb)*32 + lane)*16 + i holds, per ISA 16-bit A 16x32 map:
//   grp=lane>>4, lid=lane&15, r=i>>1, h=i&1
//   m = tm*16+lid, k = kb*32 + ((r<4)? 2r+h : 16+2(r-4)+h) + 8*grp
// Each lane's 16 bf16 are contiguous -> GEMM loads them as one v16bf (b128 x2)
// ---------------------------------------------------------------------------
__global__ void k_pack_a(const float* __restrict__ X, __bf16* __restrict__ dst,
                         int M, int K)
{
  int t = blockIdx.x * blockDim.x + threadIdx.x;
  if (t >= M * K) return;
  int i = t & 15, lane = (t >> 4) & 31, blk = t >> 9;
  int KB = K >> 5;
  int kb = blk % KB, tm = blk / KB;
  int grp = lane >> 4, lid = lane & 15;
  int r = i >> 1, h = i & 1;
  int ka = ((r < 4) ? (2 * r + h) : (16 + 2 * (r - 4) + h)) + 8 * grp;
  int m = tm * 16 + lid;
  int k = kb * 32 + ka;
  dst[t] = f2bf(X[(size_t)m * K + k]);
}

// ---------------------------------------------------------------------------
// Pack f32 row-major W[K,Ncol] into WMMA B-fragment order, bf16.
// Layout: ((tn*KB + kb)*32 + lane)*16 + i with
//   k = kb*32 + ((r<4)? 2r+h : 16+2(r-4)+h) + 8*grp,  n = tn*16+lid
// ---------------------------------------------------------------------------
__global__ void k_pack_b(const float* __restrict__ W, __bf16* __restrict__ dst,
                         int K, int Ncol)
{
  int t = blockIdx.x * blockDim.x + threadIdx.x;
  if (t >= K * Ncol) return;
  int i = t & 15, lane = (t >> 4) & 31, blk = t >> 9;
  int KB = K >> 5;
  int kb = blk % KB, tn = blk / KB;
  int grp = lane >> 4, lid = lane & 15;
  int r = i >> 1, h = i & 1;
  int k = kb * 32 + ((r < 4) ? (2 * r + h) : (16 + 2 * (r - 4) + h)) + 8 * grp;
  int n = tn * 16 + lid;
  dst[t] = f2bf(W[(size_t)k * Ncol + n]);
}

// ---------------------------------------------------------------------------
// WMMA GEMM on pre-packed bf16 fragments. One wave per 16x64 output strip
// (4 N-tiles share one A fragment -> 4 v_wmma per 32-deep k-step).
// Requires M%16==0, K%32==0, Ncol%64==0.
// ---------------------------------------------------------------------------
__global__ void k_gemm_wmma(const __bf16* __restrict__ Apack,
                            const __bf16* __restrict__ Bpack,
                            const float* __restrict__ bias, float* __restrict__ Out,
                            int Kdim, int Ncol, int relu)
{
  const int lane = threadIdx.x & 31;
  const int grp  = lane >> 4;
  const int lid  = lane & 15;
  const int tm   = blockIdx.y;
  const int tn0  = blockIdx.x * 4;
  const int KB   = Kdim >> 5;

  v8f acc0 = {}, acc1 = {}, acc2 = {}, acc3 = {};
  const __bf16* ap = Apack + ((size_t)tm * KB * 32 + lane) * 16;
  const __bf16* bp = Bpack + (size_t)lane * 16;
  const size_t bstride = (size_t)KB * 512;   // elems per N-tile of Bpack

  for (int kb = 0; kb < KB; ++kb) {
    v16bf a  = *(const v16bf*)(ap + (size_t)kb * 512);
    const __bf16* bk = bp + (size_t)kb * 512;
    v16bf b0 = *(const v16bf*)(bk + (size_t)(tn0 + 0) * bstride);
    v16bf b1 = *(const v16bf*)(bk + (size_t)(tn0 + 1) * bstride);
    v16bf b2 = *(const v16bf*)(bk + (size_t)(tn0 + 2) * bstride);
    v16bf b3 = *(const v16bf*)(bk + (size_t)(tn0 + 3) * bstride);
    acc0 = __builtin_amdgcn_wmma_f32_16x16x32_bf16(false, a, false, b0, (short)0, acc0, false, false);
    acc1 = __builtin_amdgcn_wmma_f32_16x16x32_bf16(false, a, false, b1, (short)0, acc1, false, false);
    acc2 = __builtin_amdgcn_wmma_f32_16x16x32_bf16(false, a, false, b2, (short)0, acc2, false, false);
    acc3 = __builtin_amdgcn_wmma_f32_16x16x32_bf16(false, a, false, b3, (short)0, acc3, false, false);
  }

#pragma unroll
  for (int j = 0; j < 4; ++j) {
    const v8f acc = (j == 0) ? acc0 : (j == 1) ? acc1 : (j == 2) ? acc2 : acc3;
    const int n = (tn0 + j) * 16 + lid;
    const float bv = bias[n];
#pragma unroll
    for (int r = 0; r < 8; ++r) {     // C/D: VGPR r -> M = r + 8*grp, N across lanes
      int m = tm * 16 + r + 8 * grp;
      float v = acc[r] + bv;
      if (relu) v = fmaxf(v, 0.0f);
      Out[(size_t)m * Ncol + n] = v;
    }
  }
}

// ---------------------------------------------------------------------------
// P[n, h*16+j] = sum_x Q[n, h*64+x] * We[j*256 + h*64 + x]
// (q . e  ==  ea . P[dst]) -- removes the 16x64 MAC from the edge hot loop.
// ---------------------------------------------------------------------------
__global__ void k_qproj(const float* __restrict__ Q, const float* __restrict__ We,
                        float* __restrict__ P)
{
  int t = blockIdx.x * blockDim.x + threadIdx.x;
  if (t >= N_NODES * HID) return;
  int n = t >> 6, hj = t & 63;
  int h = hj >> 4, j = hj & 15;
  const float4* qp = (const float4*)(Q + (size_t)n * HD + h * D_DIM);
  const float4* wp = (const float4*)(We + (size_t)j * HD + h * D_DIM);
  float acc = 0.0f;
#pragma unroll
  for (int i = 0; i < 16; ++i) {
    float4 a = qp[i], b = wp[i];
    acc += a.x * b.x + a.y * b.y + a.z * b.z + a.w * b.w;
  }
  P[t] = acc;
}

// ---------------------------------------------------------------------------
// Edge pass A: logit = (q.k + ea.P[dst]) * SCALE; segment-max via atomicMax.
// ---------------------------------------------------------------------------
__global__ void k_edge_logits(const float* __restrict__ Q, const float* __restrict__ Kb,
                              const float* __restrict__ EA, const float* __restrict__ P,
                              const int* __restrict__ src, const int* __restrict__ dst,
                              float* __restrict__ logits, unsigned* __restrict__ mmax)
{
  int t = blockIdx.x * blockDim.x + threadIdx.x;
  if (t >= N_EDGES * H_HEADS) return;
  int e = t >> 2, h = t & 3;
  int s = src[e], d = dst[e];
  const float4* qp = (const float4*)(Q  + (size_t)d * HD + h * D_DIM);
  const float4* kp = (const float4*)(Kb + (size_t)s * HD + h * D_DIM);
  float acc = 0.0f;
#pragma unroll
  for (int i = 0; i < 16; ++i) {
    float4 a = qp[i], b = kp[i];
    acc += a.x * b.x + a.y * b.y + a.z * b.z + a.w * b.w;
  }
  const float4* ea = (const float4*)(EA + (size_t)e * EDGE_DIM);
  const float4* pd = (const float4*)(P + (size_t)d * HID + h * 16);
  float pe = 0.0f;
#pragma unroll
  for (int i = 0; i < 4; ++i) {
    float4 a = ea[i], b = pd[i];
    pe += a.x * b.x + a.y * b.y + a.z * b.z + a.w * b.w;
  }
  float lg = (acc + pe) * SCALE_ATT;
  logits[t] = lg;
  atomicMax(&mmax[(size_t)d * H_HEADS + h], fenc(lg));
}

// ---------------------------------------------------------------------------
// Edge pass B: a = exp(logit - m[dst]); segment-sum via atomicAdd; store a.
// ---------------------------------------------------------------------------
__global__ void k_edge_expsum(float* __restrict__ logits, const unsigned* __restrict__ mmax,
                              float* __restrict__ ssum, const int* __restrict__ dst)
{
  int t = blockIdx.x * blockDim.x + threadIdx.x;
  if (t >= N_EDGES * H_HEADS) return;
  int e = t >> 2, h = t & 3;
  int d = dst[e];
  float m = fdec(mmax[(size_t)d * H_HEADS + h]);
  float a = __expf(logits[t] - m);
  logits[t] = a;
  atomicAdd(&ssum[(size_t)d * H_HEADS + h], a);
}

// ---------------------------------------------------------------------------
// Edge pass C: out[dst,h,:] += coef * v[src,h,:]  and  cea[dst,h,:] += coef*ea
// (the edge-embedding part is factored: sum_e coef*(ea@We) = (sum_e coef*ea)@We)
// ---------------------------------------------------------------------------
__global__ void k_edge_accum(const float* __restrict__ logits, const float* __restrict__ ssum,
                             const float* __restrict__ V, const float* __restrict__ EA,
                             const int* __restrict__ src, const int* __restrict__ dst,
                             float* __restrict__ outb, float* __restrict__ cea)
{
  int t = blockIdx.x * blockDim.x + threadIdx.x;
  if (t >= N_EDGES * H_HEADS) return;
  int e = t >> 2, h = t & 3;
  int s = src[e], d = dst[e];
  float coef = logits[t] / (ssum[(size_t)d * H_HEADS + h] + 1e-16f);
  const float* vp = V + (size_t)s * HD + h * D_DIM;
  float* op = outb + (size_t)d * HD + h * D_DIM;
#pragma unroll 4
  for (int xcol = 0; xcol < D_DIM; ++xcol)
    atomicAdd(&op[xcol], coef * vp[xcol]);
  const float* ea = EA + (size_t)e * EDGE_DIM;
  float* cp = cea + (size_t)d * HID + h * 16;
#pragma unroll
  for (int j = 0; j < EDGE_DIM; ++j)
    atomicAdd(&cp[j], coef * ea[j]);
}

// ---------------------------------------------------------------------------
// Node-level: outb[n,h,x] += cea[n,h,:] . We[:, h*64+x]
// ---------------------------------------------------------------------------
__global__ void k_apply_cea(const float* __restrict__ cea, const float* __restrict__ We,
                            float* __restrict__ outb)
{
  int t = blockIdx.x * blockDim.x + threadIdx.x;
  if (t >= N_NODES * HD) return;
  int n = t >> 8, hx = t & 255;
  int h = hx >> 6, xcol = hx & 63;
  const float* cp = cea + (size_t)n * HID + h * 16;
  float acc = 0.0f;
#pragma unroll
  for (int j = 0; j < EDGE_DIM; ++j)
    acc += cp[j] * We[(size_t)j * HD + h * D_DIM + xcol];
  outb[t] += acc;
}

// ---------------------------------------------------------------------------
// Node epilogue: mean over heads + skip, layernorm, +res, relu; accumulate
// hidden-state sum. One wave32 per node, 2 features per lane.
// ---------------------------------------------------------------------------
__global__ void k_node_finish(const float* __restrict__ outb, const float* __restrict__ skip,
                              const float* __restrict__ res, const float* __restrict__ ln_g,
                              const float* __restrict__ ln_b, float* __restrict__ h_out,
                              float* __restrict__ hsum)
{
  int gid  = blockIdx.x * blockDim.x + threadIdx.x;
  int n    = gid >> 5;
  int lane = threadIdx.x & 31;
  if (n >= N_NODES) return;
  const float* ob = outb + (size_t)n * HD;
  int t0 = lane, t1 = lane + 32;
  float v0 = 0.25f * (ob[t0] + ob[64 + t0] + ob[128 + t0] + ob[192 + t0]) + skip[(size_t)n * HID + t0];
  float v1 = 0.25f * (ob[t1] + ob[64 + t1] + ob[128 + t1] + ob[192 + t1]) + skip[(size_t)n * HID + t1];
  float sm = v0 + v1;
#pragma unroll
  for (int off = 16; off; off >>= 1) sm += __shfl_xor(sm, off, 32);
  float mu = sm * (1.0f / 64.0f);
  float d0 = v0 - mu, d1 = v1 - mu;
  float q = d0 * d0 + d1 * d1;
#pragma unroll
  for (int off = 16; off; off >>= 1) q += __shfl_xor(q, off, 32);
  float inv = rsqrtf(q * (1.0f / 64.0f) + 1e-5f);
  float h0 = fmaxf(d0 * inv * ln_g[t0] + ln_b[t0] + res[(size_t)n * HID + t0], 0.0f);
  float h1 = fmaxf(d1 * inv * ln_g[t1] + ln_b[t1] + res[(size_t)n * HID + t1], 0.0f);
  h_out[(size_t)n * HID + t0] = h0;
  h_out[(size_t)n * HID + t1] = h1;
  hsum[(size_t)n * HID + t0] += h0;
  hsum[(size_t)n * HID + t1] += h1;
}

// ---------------------------------------------------------------------------
// Graph pooling
// ---------------------------------------------------------------------------
__global__ void k_count_nodes(const int* __restrict__ batch, float* __restrict__ cnt) {
  int t = blockIdx.x * blockDim.x + threadIdx.x;
  if (t < N_NODES) atomicAdd(&cnt[batch[t]], 1.0f);
}

__global__ void k_pool(const float* __restrict__ hsum, const int* __restrict__ batch,
                       float* __restrict__ psum, unsigned* __restrict__ pmax)
{
  int t = blockIdx.x * blockDim.x + threadIdx.x;
  if (t >= N_NODES * HID) return;
  int n = t >> 6, d = t & 63;
  float xm = hsum[t] * 0.25f;   // mean over 4 layers
  int g = batch[n];
  atomicAdd(&psum[(size_t)g * HID + d], xm);
  atomicMax(&pmax[(size_t)g * HID + d], fenc(xm));
}

__global__ void k_pool_final(const float* __restrict__ psum, const unsigned* __restrict__ pmax,
                             const float* __restrict__ cnt, float* __restrict__ gfeat)
{
  int t = blockIdx.x * blockDim.x + threadIdx.x;
  if (t >= G_GRAPHS * 2 * HID) return;
  int g = t >> 7, d = t & 127;
  float val;
  if (d < HID) {
    val = psum[(size_t)g * HID + d] / fmaxf(cnt[g], 1.0f);
  } else {
    unsigned u = pmax[(size_t)g * HID + (d - HID)];
    val = (u == ENC_NEG_INF) ? 0.0f : fdec(u);   // isfinite guard
  }
  gfeat[t] = val;
}

// ---------------------------------------------------------------------------
// Tiny head GEMM: out[G,10] = hid[G,64] @ W2[64,10] + b2
// ---------------------------------------------------------------------------
__global__ void k_head2(const float* __restrict__ hid, const float* __restrict__ W2,
                        const float* __restrict__ b2, float* __restrict__ out)
{
  int t = blockIdx.x * blockDim.x + threadIdx.x;
  if (t >= G_GRAPHS * N_CLASSES) return;
  int g = t / N_CLASSES, c = t % N_CLASSES;
  float acc = b2[c];
  for (int j = 0; j < HID; ++j) acc += hid[(size_t)g * HID + j] * W2[j * N_CLASSES + c];
  out[t] = acc;
}

// ---------------------------------------------------------------------------
// Host side
// ---------------------------------------------------------------------------
static inline void fill_u32(unsigned* p, unsigned v, long long n, hipStream_t s) {
  int blocks = (int)((n + 255) / 256);
  k_fill_u32<<<blocks, 256, 0, s>>>(p, v, n);
}
static inline void pack_b(const float* W, __bf16* dst, int K, int Ncol, hipStream_t s) {
  int n = K * Ncol;
  k_pack_b<<<(n + 255) / 256, 256, 0, s>>>(W, dst, K, Ncol);
}

extern "C" void kernel_launch(void* const* d_in, const int* in_sizes, int n_in,
                              void* d_out, int out_size, void* d_ws, size_t ws_size,
                              hipStream_t stream) {
  (void)in_sizes; (void)n_in; (void)out_size; (void)ws_size;
  int idx = 0;
  const float* x         = (const float*)d_in[idx++];
  const float* edge_attr = (const float*)d_in[idx++];

  struct LayerP { const float *Wq,*bq,*Wk,*bk,*Wv,*bv,*We,*Wskip,*bskip,*ln_g,*ln_b,*Wres,*bres; };
  LayerP lp[4];
  for (int li = 0; li < 4; ++li) {
    lp[li].Wq    = (const float*)d_in[idx++]; lp[li].bq    = (const float*)d_in[idx++];
    lp[li].Wk    = (const float*)d_in[idx++]; lp[li].bk    = (const float*)d_in[idx++];
    lp[li].Wv    = (const float*)d_in[idx++]; lp[li].bv    = (const float*)d_in[idx++];
    lp[li].We    = (const float*)d_in[idx++];
    lp[li].Wskip = (const float*)d_in[idx++]; lp[li].bskip = (const float*)d_in[idx++];
    lp[li].ln_g  = (const float*)d_in[idx++]; lp[li].ln_b  = (const float*)d_in[idx++];
    if (li == 0) { lp[li].Wres = (const float*)d_in[idx++]; lp[li].bres = (const float*)d_in[idx++]; }
    else         { lp[li].Wres = nullptr;                   lp[li].bres = nullptr; }
  }
  const float* W1 = (const float*)d_in[idx++]; const float* b1 = (const float*)d_in[idx++];
  const float* W2 = (const float*)d_in[idx++]; const float* b2 = (const float*)d_in[idx++];
  const int* edge_index = (const int*)d_in[idx++];
  const int* batch      = (const int*)d_in[idx++];
  const int* src = edge_index;
  const int* dst = edge_index + N_EDGES;

  // ---------------- workspace layout (f32 elements; all 64B-aligned) -------
  float* ws = (float*)d_ws;
  float* buf_q    = ws; ws += (size_t)N_NODES * HD;   // Q, reused as attention-out
  float* buf_k    = ws; ws += (size_t)N_NODES * HD;
  float* buf_v    = ws; ws += (size_t)N_NODES * HD;
  float* buf_skip = ws; ws += (size_t)N_NODES * HID;
  float* buf_res  = ws; ws += (size_t)N_NODES * HID;
  float* buf_h    = ws; ws += (size_t)N_NODES * HID;
  float* buf_hsum = ws; ws += (size_t)N_NODES * HID;
  float* buf_P    = ws; ws += (size_t)N_NODES * HID;  // We.q projections
  float* buf_cea  = ws; ws += (size_t)N_NODES * HID;  // sum coef*edge_attr
  unsigned* buf_m = (unsigned*)ws; ws += (size_t)N_NODES * H_HEADS;
  float* buf_s    = ws; ws += (size_t)N_NODES * H_HEADS;
  float* buf_a    = ws; ws += (size_t)N_EDGES * H_HEADS;
  float* psum     = ws; ws += (size_t)G_GRAPHS * HID;
  unsigned* pmax  = (unsigned*)ws; ws += (size_t)G_GRAPHS * HID;
  float* pcnt     = ws; ws += (size_t)G_GRAPHS;
  float* gfeat    = ws; ws += (size_t)G_GRAPHS * 2 * HID;
  float* ghid     = ws; ws += (size_t)G_GRAPHS * HID;
  // bf16 packed fragments (32B-aligned: all offsets are multiples of 16 f32)
  __bf16* apack   = (__bf16*)ws; ws += (size_t)N_NODES * F_INPUT / 2;  // max M*K bf16
  __bf16* bpack0  = (__bf16*)ws;                                       // all packed weights

  // Pack every weight matrix into B-fragment order (bf16) once per call.
  __bf16* bp = bpack0;
  __bf16 *pWq[4], *pWk[4], *pWv[4], *pWskip[4], *pWres0, *pW1;
  for (int li = 0; li < 4; ++li) {
    int K = (li == 0) ? F_INPUT : HID;
    pWq[li] = bp;    pack_b(lp[li].Wq,    bp, K, HD,  stream); bp += (size_t)K * HD;
    pWk[li] = bp;    pack_b(lp[li].Wk,    bp, K, HD,  stream); bp += (size_t)K * HD;
    pWv[li] = bp;    pack_b(lp[li].Wv,    bp, K, HD,  stream); bp += (size_t)K * HD;
    pWskip[li] = bp; pack_b(lp[li].Wskip, bp, K, HID, stream); bp += (size_t)K * HID;
  }
  pWres0 = bp; pack_b(lp[0].Wres, bp, F_INPUT, HID, stream); bp += (size_t)F_INPUT * HID;
  pW1    = bp; pack_b(W1, bp, 2 * HID, HID, stream);         bp += (size_t)2 * HID * HID;

  const int EH = N_EDGES * H_HEADS;
  const int eh_blocks = (EH + 255) / 256;

  fill_u32((unsigned*)buf_hsum, 0u, (long long)N_NODES * HID, stream);

  const float* xin = x;
  int in_dim = F_INPUT;
  for (int li = 0; li < 4; ++li) {
    const LayerP& p = lp[li];
    // Pack activations once; shared by Q/K/V/skip(/res) GEMMs.
    k_pack_a<<<(N_NODES * in_dim + 255) / 256, 256, 0, stream>>>(xin, apack, N_NODES, in_dim);

    dim3 gqkv(HD / 64, N_NODES / 16);     // 4 N-tiles per wave
    k_gemm_wmma<<<gqkv, 32, 0, stream>>>(apack, pWq[li], p.bq, buf_q, in_dim, HD, 0);
    k_gemm_wmma<<<gqkv, 32, 0, stream>>>(apack, pWk[li], p.bk, buf_k, in_dim, HD, 0);
    k_gemm_wmma<<<gqkv, 32, 0, stream>>>(apack, pWv[li], p.bv, buf_v, in_dim, HD, 0);
    dim3 gskip(HID / 64, N_NODES / 16);
    k_gemm_wmma<<<gskip, 32, 0, stream>>>(apack, pWskip[li], p.bskip, buf_skip, in_dim, HID, 0);
    const float* res = xin;
    if (li == 0) {
      k_gemm_wmma<<<gskip, 32, 0, stream>>>(apack, pWres0, p.bres, buf_res, in_dim, HID, 0);
      res = buf_res;
    }

    // P = (We_h . q_n) so edges only need a 16-dim dot for the embed term.
    k_qproj<<<(N_NODES * HID + 255) / 256, 256, 0, stream>>>(buf_q, p.We, buf_P);

    fill_u32(buf_m, ENC_NEG_INF, (long long)N_NODES * H_HEADS, stream);
    fill_u32((unsigned*)buf_s, 0u, (long long)N_NODES * H_HEADS, stream);

    k_edge_logits<<<eh_blocks, 256, 0, stream>>>(buf_q, buf_k, edge_attr, buf_P, src, dst, buf_a, buf_m);
    k_edge_expsum<<<eh_blocks, 256, 0, stream>>>(buf_a, buf_m, buf_s, dst);

    // Q no longer needed -> reuse as attention-output accumulator.
    fill_u32((unsigned*)buf_q, 0u, (long long)N_NODES * HD, stream);
    fill_u32((unsigned*)buf_cea, 0u, (long long)N_NODES * HID, stream);
    k_edge_accum<<<eh_blocks, 256, 0, stream>>>(buf_a, buf_s, buf_v, edge_attr, src, dst, buf_q, buf_cea);
    k_apply_cea<<<(N_NODES * HD + 255) / 256, 256, 0, stream>>>(buf_cea, p.We, buf_q);

    int nf_blocks = (N_NODES * 32 + 255) / 256;
    k_node_finish<<<nf_blocks, 256, 0, stream>>>(buf_q, buf_skip, res, p.ln_g, p.ln_b, buf_h, buf_hsum);

    xin = buf_h;
    in_dim = HID;
  }

  // Pooling
  fill_u32((unsigned*)psum, 0u, (long long)G_GRAPHS * HID, stream);
  fill_u32(pmax, ENC_NEG_INF, (long long)G_GRAPHS * HID, stream);
  fill_u32((unsigned*)pcnt, 0u, (long long)G_GRAPHS, stream);
  k_count_nodes<<<(N_NODES + 255) / 256, 256, 0, stream>>>(batch, pcnt);
  k_pool<<<(N_NODES * HID + 255) / 256, 256, 0, stream>>>(buf_hsum, batch, psum, pmax);
  k_pool_final<<<(G_GRAPHS * 2 * HID + 255) / 256, 256, 0, stream>>>(psum, pmax, pcnt, gfeat);

  // Head: relu(g @ W1 + b1) @ W2 + b2
  k_pack_a<<<(G_GRAPHS * 2 * HID + 255) / 256, 256, 0, stream>>>(gfeat, apack, G_GRAPHS, 2 * HID);
  dim3 ghead(HID / 64, G_GRAPHS / 16);
  k_gemm_wmma<<<ghead, 32, 0, stream>>>(apack, pW1, b1, ghid, 2 * HID, HID, 1);
  k_head2<<<(G_GRAPHS * N_CLASSES + 255) / 256, 256, 0, stream>>>(ghid, W2, b2, (float*)d_out);
}
